// Discriminator_53214644797506
// MI455X (gfx1250) — matrix operand order repaired and checked by reading, not compiled, not run on hardware
//
#include <hip/hip_runtime.h>
#include <hip/hip_fp16.h>

typedef __attribute__((ext_vector_type(16))) _Float16 v16h;
typedef __attribute__((ext_vector_type(8)))  float    v8f;

#define THREADS 256
#define STAGE_SLICES 32   // 32 slices * 256 f16 = 16 KB LDS weight stage

// Plain-POD 16B / 32B chunks
struct alignas(16) H8  { unsigned int w[4]; };
struct alignas(16) H16 { H8 lo, hi; };

static __device__ __forceinline__ v16h as_v16h(const H16& x) {
    return __builtin_bit_cast(v16h, x);
}

// ---------------------------------------------------------------------------
// Weight repack: OIHW f32 -> [ct][slice][co16][ci16] f16,
// slice s = r*C16 + c16  (r = ky*3+kx), padded slices are zero.
// ---------------------------------------------------------------------------
__global__ void repack_weights_kernel(const float* __restrict__ w, __half* __restrict__ wpk,
                                      int Cin, int S, int Spad, int total)
{
    int idx = blockIdx.x * THREADS + threadIdx.x;
    if (idx >= total) return;
    int ci = idx & 15;
    int co = (idx >> 4) & 15;
    int s  = (idx >> 8) % Spad;
    int ct = (idx >> 8) / Spad;
    int C16 = (Cin + 15) >> 4;
    float v = 0.0f;
    if (s < S) {
        int r   = s / C16;
        int c16 = s - r * C16;
        int cig = c16 * 16 + ci;
        int cog = ct * 16 + co;
        if (cig < Cin)
            v = w[((size_t)cog * Cin + cig) * 9 + r];
    }
    wpk[idx] = __float2half(v);
}

// ---------------------------------------------------------------------------
// Build h0 = [x | t @ w_t^T + b_t] as NHWC f16, Cpad=16 (ch 0 = x, 1..8 = tt)
// ---------------------------------------------------------------------------
__global__ void build_h0_kernel(const float* __restrict__ x, const float* __restrict__ t,
                                const float* __restrict__ w_t, const float* __restrict__ b_t,
                                __half* __restrict__ h0)
{
    int idx = blockIdx.x * THREADS + threadIdx.x;   // n*1024 + p
    if (idx >= 1024 * 1024) return;
    int n = idx >> 10;
    int p = idx & 1023;
    float tv[10];
#pragma unroll
    for (int k = 0; k < 10; ++k) tv[k] = t[n * 10 + k];
    __align__(16) __half o[16];
    o[0] = __float2half(x[idx]);
#pragma unroll
    for (int j = 0; j < 8; ++j) {
        int row = j * 1024 + p;
        float acc = b_t[row];
#pragma unroll
        for (int k = 0; k < 10; ++k) acc += tv[k] * w_t[(size_t)row * 10 + k];
        o[1 + j] = __float2half(acc);
    }
#pragma unroll
    for (int c = 9; c < 16; ++c) o[c] = __float2half(0.0f);
    H8* dst = (H8*)(h0 + (size_t)idx * 16);
    dst[0] = ((const H8*)o)[0];
    dst[1] = ((const H8*)o)[1];
}

// ---------------------------------------------------------------------------
// Message passing: per (node, pixel) accumulate signed neighbor sums (f32),
// write ngh (sign>0) and non (sign<0) as NHWC f16 (16 ch). P is a power of 2.
// ---------------------------------------------------------------------------
__global__ void mpn_gather_kernel(const __half* __restrict__ xact, const int* __restrict__ edges,
                                  __half* __restrict__ ngh, __half* __restrict__ non,
                                  int P, int pShift)
{
    int idx = blockIdx.x * THREADS + threadIdx.x;   // node*P + pos
    if (idx >= 1024 * P) return;
    int node = idx >> pShift;
    int pos  = idx & (P - 1);
    int g = node >> 4;
    float ap[16], an[16];
#pragma unroll
    for (int i = 0; i < 16; ++i) { ap[i] = 0.0f; an[i] = 0.0f; }
    const int* eg = edges + (size_t)g * 120 * 3;
    for (int e = 0; e < 120; ++e) {
        int a = eg[e * 3 + 0];
        int s = eg[e * 3 + 1];
        int b = eg[e * 3 + 2];
        int partner = (a == node) ? b : ((b == node) ? a : -1);
        if (partner >= 0) {
            const H8* sp = (const H8*)(xact + (((size_t)partner << pShift) + pos) * 16);
            __align__(16) __half vals[16];
            ((H8*)vals)[0] = sp[0];
            ((H8*)vals)[1] = sp[1];
            if (s > 0) {
#pragma unroll
                for (int i = 0; i < 16; ++i) ap[i] += __half2float(vals[i]);
            } else {
#pragma unroll
                for (int i = 0; i < 16; ++i) an[i] += __half2float(vals[i]);
            }
        }
    }
    __align__(16) __half op[16], on[16];
#pragma unroll
    for (int i = 0; i < 16; ++i) { op[i] = __float2half(ap[i]); on[i] = __float2half(an[i]); }
    H8* dp = (H8*)(ngh + (size_t)idx * 16);
    dp[0] = ((const H8*)op)[0]; dp[1] = ((const H8*)op)[1];
    H8* dn = (H8*)(non + (size_t)idx * 16);
    dn[0] = ((const H8*)on)[0]; dn[1] = ((const H8*)on)[1];
}

// ---------------------------------------------------------------------------
// Implicit-GEMM 3x3 conv (pad=1) via V_WMMA_F32_16X16X32_F16.
// Per wave: TWO 16(cout) x 16(pixels) tiles sharing one A (weight) fragment ->
// two independent accumulators; K-loop unrolled 2x -> 4 independent v_wmma
// in flight. Weights staged into LDS with GLOBAL_LOAD_ASYNC_TO_LDS_B128
// (ASYNCcnt) and shared by all 8 waves; activations streamed NHWC from L2.
// Grid: (posBlocks, coutTiles, N).
// ---------------------------------------------------------------------------
__global__ __launch_bounds__(THREADS) void conv_wmma_kernel(
    const __half* __restrict__ inA, const __half* __restrict__ inB, const __half* __restrict__ inC,
    int nsrc, int cpadin,
    const __half* __restrict__ wpk, const float* __restrict__ bias,
    __half* __restrict__ outp, int cpadout,
    int H, int W, int HW, int Wo, int woShift, int P, int stride,
    int C16, int Spad, int lrelu)
{
    __shared__ __align__(16) __half wlds[STAGE_SLICES * 256];

    int n  = blockIdx.z;
    int ct = blockIdx.y;
    int pb = blockIdx.x;

    int wave = threadIdx.x >> 5;
    int lane = threadIdx.x & 31;
    int hi   = lane >> 4;     // 0: K 0..15 / M 0..7 ; 1: K 16..31 / M 8..15
    int co   = lane & 15;     // also pixel index within a tile
    int ci0  = hi * 8;

    int tileA = pb * 16 + wave;
    int tileB = tileA + 8;
    int lanePosA = tileA * 16 + co;
    int lanePosB = tileB * 16 + co;
    bool validA = lanePosA < P;
    bool validB = lanePosB < P;
    int lpA = validA ? lanePosA : (P - 1);
    int lpB = validB ? lanePosB : (P - 1);
    int oyA = (lpA >> woShift) * stride, oxA = (lpA & (Wo - 1)) * stride;
    int oyB = (lpB >> woShift) * stride, oxB = (lpB & (Wo - 1)) * stride;

    v8f accA = {};
    v8f accB = {};

    const __half* wsrc = wpk + (size_t)ct * Spad * 256;
    const H8 zero8 = {{0u, 0u, 0u, 0u}};

    int ky = 0, kx = 0, cc = 0;   // coords of current even slice

    for (int sb = 0; sb < Spad; sb += STAGE_SLICES) {
        int cnt = Spad - sb; if (cnt > STAGE_SLICES) cnt = STAGE_SLICES;
        __syncthreads();
        {
            // Async DMA of the weight stage straight into LDS (no VGPR round
            // trip): each active lane copies 16 B; tracked with ASYNCcnt.
            const H8* src = (const H8*)(wsrc + (size_t)sb * 256);
            for (int i = threadIdx.x; i < cnt * 32; i += THREADS) {
                unsigned           ldsOff = (unsigned)i * 16u;   // wlds is at LDS offset 0
                unsigned long long gaddr  = (unsigned long long)(const void*)(src + i);
                asm volatile("global_load_async_to_lds_b128 %0, %1, off"
                             :: "v"(ldsOff), "v"(gaddr)
                             : "memory");
            }
            asm volatile("s_wait_asynccnt 0x0" ::: "memory");
        }
        if (sb + STAGE_SLICES < Spad)   // prefetch next weight stage (global_prefetch_b8)
            __builtin_prefetch(wsrc + (size_t)(sb + STAGE_SLICES) * 256 + threadIdx.x * 32, 0, 1);
        __syncthreads();

#pragma unroll 2
        for (int s2 = 0; s2 < cnt; s2 += 2) {
            // A fragment: w[slice s2+hi][co][0..7] (v0-3) | w[slice s2+hi][co][8..15] (v4-7)
            H16 ha;
            ha.lo = *(const H8*)(wlds + (s2 + hi) * 256 + co * 16);
            ha.hi = *(const H8*)(wlds + (s2 + hi) * 256 + co * 16 + 8);

            // odd-slice coords (incremental advance, no division)
            int ky1 = ky, kx1 = kx, cc1 = cc + 1;
            if (cc1 == C16) { cc1 = 0; kx1 = kx + 1; if (kx1 == 3) { kx1 = 0; ky1 = ky + 1; } }

            // this lane's B slice
            int kyl = hi ? ky1 : ky;
            int kxl = hi ? kx1 : kx;
            int ccl = hi ? cc1 : cc;
            bool slcValid = kyl < 3;   // padded slices have ky >= 3

            const __half* base; int cp, coff;
            if (nsrc == 3) {
                base = (ccl == 0) ? inA : ((ccl == 1) ? inB : inC);
                cp = 16; coff = 0;
            } else {
                base = inA; cp = cpadin; coff = ccl * 16;
            }

            H16 hbA; hbA.lo = zero8; hbA.hi = zero8;
            H16 hbB; hbB.lo = zero8; hbB.hi = zero8;
            {
                int iy = oyA + kyl - 1;
                int ix = oxA + kxl - 1;
                if (slcValid && iy >= 0 && iy < H && ix >= 0 && ix < W) {
                    const H8* sp = (const H8*)(base + ((size_t)n * HW + iy * W + ix) * cp + coff);
                    hbA.lo = sp[0]; hbA.hi = sp[1];
                }
            }
            {
                int iy = oyB + kyl - 1;
                int ix = oxB + kxl - 1;
                if (slcValid && iy >= 0 && iy < H && ix >= 0 && ix < W) {
                    const H8* sp = (const H8*)(base + ((size_t)n * HW + iy * W + ix) * cp + coff);
                    hbB.lo = sp[0]; hbB.hi = sp[1];
                }
            }

            v16h a = as_v16h(ha);
            accA = __builtin_amdgcn_wmma_f32_16x16x32_f16(
                       false, a, false, as_v16h(hbA), (short)0, accA, false, false);
            accB = __builtin_amdgcn_wmma_f32_16x16x32_f16(
                       false, a, false, as_v16h(hbB), (short)0, accB, false, false);

            // advance even coords by two slices
            cc = cc1 + 1; kx = kx1; ky = ky1;
            if (cc == C16) { cc = 0; ++kx; if (kx == 3) { kx = 0; ++ky; } }
        }
    }

    float bv[8];
#pragma unroll
    for (int i = 0; i < 8; ++i) bv[i] = bias[ct * 16 + ci0 + i];

    if (validA) {
        __align__(16) __half o8[8];
#pragma unroll
        for (int i = 0; i < 8; ++i) {
            float v = accA[i] + bv[i];                 // M = i + 8*hi
            if (lrelu) v = (v > 0.0f) ? v : 0.1f * v;
            o8[i] = __float2half(v);
        }
        *(H8*)(outp + ((size_t)n * P + lpA) * cpadout + ct * 16 + ci0) = *(const H8*)o8;
    }
    if (validB) {
        __align__(16) __half o8[8];
#pragma unroll
        for (int i = 0; i < 8; ++i) {
            float v = accB[i] + bv[i];
            if (lrelu) v = (v > 0.0f) ? v : 0.1f * v;
            o8[i] = __float2half(v);
        }
        *(H8*)(outp + ((size_t)n * P + lpB) * cpadout + ct * 16 + ci0) = *(const H8*)o8;
    }
}

// ---------------------------------------------------------------------------
// Final: pooled[g] = sum over 16 nodes of d3[n][128]; out[g] = pooled . pw + pb
// ---------------------------------------------------------------------------
__global__ void final_pool_kernel(const __half* __restrict__ d3, const float* __restrict__ pw,
                                  const float* __restrict__ pb, float* __restrict__ out)
{
    int g = blockIdx.x;
    int c = threadIdx.x;   // 0..127
    float s = 0.0f;
    for (int k = 0; k < 16; ++k)
        s += __half2float(d3[((size_t)(g * 16 + k)) * 128 + c]);
    s *= pw[c];
    __shared__ float red[128];
    red[c] = s;
    __syncthreads();
    for (int off = 64; off > 0; off >>= 1) {
        if (c < off) red[c] += red[c + off];
        __syncthreads();
    }
    if (c == 0) out[g] = red[0] + pb[0];
}

// ---------------------------------------------------------------------------
// Host orchestration
// ---------------------------------------------------------------------------
static int ilog2(int v) { int s = 0; while ((1 << s) < v) ++s; return s; }

static void run_conv(hipStream_t stream,
                     const __half* inA, const __half* inB, const __half* inC, int nsrc, int cpadin,
                     const float* w, const float* bias, __half* wpk,
                     __half* outp, int cpadout,
                     int H, int W, int Cin, int Cout, int stride, int lrelu)
{
    const int N = 1024;
    int C16  = (Cin + 15) / 16;
    int S    = 9 * C16;
    int Spad = (S + 1) & ~1;
    int ct   = Cout / 16;
    int totalW = ct * Spad * 256;
    repack_weights_kernel<<<(totalW + THREADS - 1) / THREADS, THREADS, 0, stream>>>(
        w, wpk, Cin, S, Spad, totalW);
    int Ho = (H - 1) / stride + 1;
    int Wo = (W - 1) / stride + 1;
    int P  = Ho * Wo;
    int posTiles = (P + 15) / 16;
    int bpic = (posTiles + 15) / 16;       // 16 tiles per block (2 per wave)
    dim3 grid((unsigned)bpic, (unsigned)ct, (unsigned)N);
    conv_wmma_kernel<<<grid, THREADS, 0, stream>>>(
        inA, inB, inC, nsrc, cpadin, wpk, bias, outp, cpadout,
        H, W, H * W, Wo, ilog2(Wo), P, stride, C16, Spad, lrelu);
}

extern "C" void kernel_launch(void* const* d_in, const int* in_sizes, int n_in,
                              void* d_out, int out_size, void* d_ws, size_t ws_size,
                              hipStream_t stream)
{
    (void)in_sizes; (void)n_in; (void)out_size; (void)ws_size;
    const float* x     = (const float*)d_in[0];
    const float* t     = (const float*)d_in[1];
    const int*   edges = (const int*)d_in[2];
    const float* w_t   = (const float*)d_in[4];
    const float* b_t   = (const float*)d_in[5];
    const float* c1w  = (const float*)d_in[6];  const float* c1b  = (const float*)d_in[7];
    const float* c2w  = (const float*)d_in[8];  const float* c2b  = (const float*)d_in[9];
    const float* c3w  = (const float*)d_in[10]; const float* c3b  = (const float*)d_in[11];
    const float* m1c1w= (const float*)d_in[12]; const float* m1c1b= (const float*)d_in[13];
    const float* m1c2w= (const float*)d_in[14]; const float* m1c2b= (const float*)d_in[15];
    const float* m1c3w= (const float*)d_in[16]; const float* m1c3b= (const float*)d_in[17];
    const float* ds1w = (const float*)d_in[18]; const float* ds1b = (const float*)d_in[19];
    const float* m2c1w= (const float*)d_in[20]; const float* m2c1b= (const float*)d_in[21];
    const float* m2c2w= (const float*)d_in[22]; const float* m2c2b= (const float*)d_in[23];
    const float* m2c3w= (const float*)d_in[24]; const float* m2c3b= (const float*)d_in[25];
    const float* ds2w = (const float*)d_in[26]; const float* ds2b = (const float*)d_in[27];
    const float* d1w  = (const float*)d_in[28]; const float* d1b  = (const float*)d_in[29];
    const float* d2w  = (const float*)d_in[30]; const float* d2b  = (const float*)d_in[31];
    const float* d3w  = (const float*)d_in[32]; const float* d3b  = (const float*)d_in[33];
    const float* pw   = (const float*)d_in[34]; const float* pb   = (const float*)d_in[35];

    __half* ws = (__half*)d_ws;
    __half* W0 = ws;                           // 64 MB slot
    __half* W1 = ws + ((size_t)32 << 20);      // 64 MB slot
    __half* W2 = ws + ((size_t)64 << 20);      // 32 MB slot
    __half* W3 = ws + ((size_t)80 << 20);      // 32 MB slot
    __half* wpcur = ws + ((size_t)96 << 20);   // packed-weight arena (~1.1 MB)

    auto take_wpk = [&](int Cin, int Cout) -> __half* {
        int C16  = (Cin + 15) / 16;
        int Spad = (9 * C16 + 1) & ~1;
        __half* r = wpcur;
        wpcur += (size_t)(Cout / 16) * Spad * 256;
        return r;
    };

    // h0 (x | t-embedding), NHWC f16 Cpad=16            -> W0
    build_h0_kernel<<<4096, THREADS, 0, stream>>>(x, t, w_t, b_t, W0);
    // encoder convs (lrelu)
    run_conv(stream, W0, nullptr, nullptr, 1, 16, c1w, c1b, take_wpk(9, 16),  W1, 16, 32, 32, 9,  16, 1, 1);
    run_conv(stream, W1, nullptr, nullptr, 1, 16, c2w, c2b, take_wpk(16, 16), W0, 16, 32, 32, 16, 16, 1, 1);
    run_conv(stream, W0, nullptr, nullptr, 1, 16, c3w, c3b, take_wpk(16, 16), W1, 16, 32, 32, 16, 16, 1, 1);
    // MPN 1 @ 32x32: gather then 48->32->32->16 (no activation)
    mpn_gather_kernel<<<(1024 * 1024 + THREADS - 1) / THREADS, THREADS, 0, stream>>>(
        W1, edges, W2, W3, 1024, 10);
    run_conv(stream, W1, W2, W3, 3, 16, m1c1w, m1c1b, take_wpk(48, 32), W0, 32, 32, 32, 48, 32, 1, 0);
    run_conv(stream, W0, nullptr, nullptr, 1, 32, m1c2w, m1c2b, take_wpk(32, 32), W1, 32, 32, 32, 32, 32, 1, 0);
    run_conv(stream, W1, nullptr, nullptr, 1, 32, m1c3w, m1c3b, take_wpk(32, 16), W2, 16, 32, 32, 32, 16, 1, 0);
    // downsample 1 (stride 2, lrelu) -> 16x16
    run_conv(stream, W2, nullptr, nullptr, 1, 16, ds1w, ds1b, take_wpk(16, 16), W3, 16, 32, 32, 16, 16, 2, 1);
    // MPN 2 @ 16x16
    __half* ds1o = W3;
    __half* ngh2 = W3 + ((size_t)4 << 20);
    __half* non2 = W3 + ((size_t)8 << 20);
    mpn_gather_kernel<<<(1024 * 256 + THREADS - 1) / THREADS, THREADS, 0, stream>>>(
        ds1o, edges, ngh2, non2, 256, 8);
    run_conv(stream, ds1o, ngh2, non2, 3, 16, m2c1w, m2c1b, take_wpk(48, 32), W0, 32, 16, 16, 48, 32, 1, 0);
    run_conv(stream, W0, nullptr, nullptr, 1, 32, m2c2w, m2c2b, take_wpk(32, 32), W1, 32, 16, 16, 32, 32, 1, 0);
    run_conv(stream, W1, nullptr, nullptr, 1, 32, m2c3w, m2c3b, take_wpk(32, 16), W2, 16, 16, 16, 32, 16, 1, 0);
    // downsample 2 (stride 2, lrelu) -> 8x8
    run_conv(stream, W2, nullptr, nullptr, 1, 16, ds2w, ds2b, take_wpk(16, 16), W0, 16, 16, 16, 16, 16, 2, 1);
    // head: 16->256 (4x4), 256->128 (2x2), 128->128 (1x1), all stride2 + lrelu
    run_conv(stream, W0, nullptr, nullptr, 1, 16,  d1w, d1b, take_wpk(16, 256),  W1, 256, 8, 8, 16, 256, 2, 1);
    run_conv(stream, W1, nullptr, nullptr, 1, 256, d2w, d2b, take_wpk(256, 128), W2, 128, 4, 4, 256, 128, 2, 1);
    run_conv(stream, W2, nullptr, nullptr, 1, 128, d3w, d3b, take_wpk(128, 128), W3, 128, 2, 2, 128, 128, 2, 1);
    // pooled dot product -> d_out (64 floats)
    final_pool_kernel<<<64, 128, 0, stream>>>(W3, pw, pb, (float*)d_out);
}
